// EdisonPerceiverAutoEncoder_14396730376860
// MI455X (gfx1250) — matrix operand order repaired and checked by reading, not compiled, not run on hardware
//
#include <hip/hip_runtime.h>
#include <cstdint>
#include <cstddef>

// ============================================================================
// EdisonPerceiverAutoEncoder forward for MI455X (gfx1250, wave32, WMMA).
//
// All large GEMMs (~94% of FLOPs) run through v_wmma_f32_16x16x32_bf16 with
// f32 accumulation; activations/weights staged as bf16 (weights ~N(0,0.02)).
// GEMM: 128x128 block tile, BK=64, double-buffered LDS, 8 waves x (2x4)
// 16x16 WMMA tiles -> 16 WMMAs per barrier. Tiles are moved global->LDS with
// GLOBAL_LOAD_ASYNC_TO_LDS_B128 (ASYNCcnt) when the toolchain exposes the
// builtin (zero VGPR staging -> no spills); otherwise a register-staged
// fallback with a relaxed VGPR budget (__launch_bounds__(256,1)).
// LN / RMS / softmax / attention run in f32 VALU (tiny FLOP share).
//
// Input flattening assumption: jax.tree-style flatten of setup_inputs() dict
// (dict keys sorted recursively) -> 43 leaves, mapped below.
// Workspace budget: ~450 MB (bf16 weight copies + activations).
// ============================================================================

typedef __bf16 bf16_t;
typedef __attribute__((ext_vector_type(16))) __bf16 v16bf;
typedef __attribute__((ext_vector_type(8)))  float  v8f;

#if __has_builtin(__builtin_amdgcn_global_load_async_to_lds_b128) && \
    __has_builtin(__builtin_amdgcn_s_wait_asynccnt)
#define USE_ASYNC_LDS 1
// builtin signature (from clang diagnostics): non-const v4i* in global (AS1)
// and LDS (AS3) address spaces.
typedef int v4i_vs __attribute__((__vector_size__(16)));
#define GAS_CAST(p) ((__attribute__((address_space(1))) v4i_vs*)(p))
#define LAS_CAST(p) ((__attribute__((address_space(3))) v4i_vs*)(p))
#else
#define USE_ASYNC_LDS 0
#endif

// ---------------------------------------------------------------------------
// Elementwise helpers
// ---------------------------------------------------------------------------
__global__ void k_f32_to_bf16(const float* __restrict__ s, bf16_t* __restrict__ d, size_t n) {
  size_t i = (size_t)blockIdx.x * blockDim.x + threadIdx.x;
  size_t stride = (size_t)gridDim.x * blockDim.x;
  for (; i < n; i += stride) d[i] = (bf16_t)s[i];
}

// out[i] = in[i] + pos[i % period] * scale   (in may alias out)
__global__ void k_add_pos(const float* __restrict__ in, const float* __restrict__ pos,
                          float* __restrict__ out, size_t n, size_t period, float scale) {
  size_t i = (size_t)blockIdx.x * blockDim.x + threadIdx.x;
  size_t stride = (size_t)gridDim.x * blockDim.x;
  for (; i < n; i += stride) out[i] = in[i] + pos[i % period] * scale;
}

// dst[i] = src[i % period]   (broadcast latents over batch)
__global__ void k_bcast(const float* __restrict__ src, float* __restrict__ dst,
                        size_t n, size_t period) {
  size_t i = (size_t)blockIdx.x * blockDim.x + threadIdx.x;
  size_t stride = (size_t)gridDim.x * blockDim.x;
  for (; i < n; i += stride) dst[i] = src[i % period];
}

// ---------------------------------------------------------------------------
// WMMA bf16 GEMM:  C(M,N) = act( A(M,K) @ B(K,N) + bias ) [+ Radd]
//  - A, B row-major bf16; accum f32 via v_wmma_f32_16x16x32_bf16
//  - block tile 128x128, BK=64, double-buffered LDS, single barrier/iter
//  - 8 waves (4x2); each wave 32x64 = 2x4 of 16x16 tiles -> 16 WMMA/iter
//  - requires M%128==0, N%128==0, K%64==0 (true for all call sites here)
// ---------------------------------------------------------------------------
#define BM 128
#define BN 128
#define BK 64
#define LDA_S 72    // bf16 elems; 144B row stride (16B multiple)
#define LDB_S 136   // bf16 elems; 272B row stride (16B multiple)
#define A_TILE (BM * LDA_S)
#define B_TILE (BK * LDB_S)

template <bool HAS_BIAS, bool GELU, bool HAS_RADD, bool OUT_F, bool OUT_B>
__global__ __launch_bounds__(256, 1)
void k_gemm(const bf16_t* __restrict__ A, const bf16_t* __restrict__ Bw,
            const float* __restrict__ bias, const float* __restrict__ Radd,
            float* __restrict__ Cf, bf16_t* __restrict__ Cb,
            int M, int N, int K) {
  __shared__ __align__(16) bf16_t As[2 * A_TILE];
  __shared__ __align__(16) bf16_t Bs[2 * B_TILE];

  const int tid  = threadIdx.x;
  const int wave = tid >> 5;
  const int lane = tid & 31;
  const int bm = blockIdx.y * BM;
  const int bn = blockIdx.x * BN;
  const int wm = (wave >> 1) * 32;   // wave row offset within block tile
  const int wn = (wave & 1) * 64;    // wave col offset within block tile

  const int lr = lane & 15;                 // A-frag: M row within 16
  const int kb = (lane < 16) ? 0 : 8;       // A-frag: K base (ISA layout)

  v8f acc[2][4] = {};

  // chunk coordinates for this thread (A: 8 chunks/row of 64; B: 16/row of 128)
  // each of 4 passes moves 16B per thread => full 128x64 A and 64x128 B tiles.

#if USE_ASYNC_LDS
  // ---- async global -> LDS copy (no VGPR data staging, ASYNCcnt) ----
  auto issueTile = [&](int k0, int buf) {
    bf16_t* a = As + buf * A_TILE;
    bf16_t* b = Bs + buf * B_TILE;
#pragma unroll
    for (int p = 0; p < 4; ++p) {
      int c = p * 256 + tid;
      int row = c >> 3, col = (c & 7) * 8;
      __builtin_amdgcn_global_load_async_to_lds_b128(
          GAS_CAST(A + (size_t)(bm + row) * K + (k0 + col)),
          LAS_CAST(&a[row * LDA_S + col]), 0, 0);
    }
#pragma unroll
    for (int p = 0; p < 4; ++p) {
      int c = p * 256 + tid;
      int row = c >> 4, col = (c & 15) * 8;
      __builtin_amdgcn_global_load_async_to_lds_b128(
          GAS_CAST(Bw + (size_t)(k0 + row) * N + (bn + col)),
          LAS_CAST(&b[row * LDB_S + col]), 0, 0);
    }
  };
  auto tileFence = [&]() {
    __builtin_amdgcn_s_wait_asynccnt(0);
    __syncthreads();
  };
#else
  // ---- fallback: register-staged copy ----
  uint4 ar[4], br[4];
  auto loadG = [&](int k0) {
#pragma unroll
    for (int p = 0; p < 4; ++p) {
      int c = p * 256 + tid;
      int row = c >> 3, col = (c & 7) * 8;
      ar[p] = *(const uint4*)(A + (size_t)(bm + row) * K + (k0 + col));
    }
#pragma unroll
    for (int p = 0; p < 4; ++p) {
      int c = p * 256 + tid;
      int row = c >> 4, col = (c & 15) * 8;
      br[p] = *(const uint4*)(Bw + (size_t)(k0 + row) * N + (bn + col));
    }
  };
  auto storeL = [&](int buf) {
    bf16_t* a = As + buf * A_TILE;
    bf16_t* b = Bs + buf * B_TILE;
#pragma unroll
    for (int p = 0; p < 4; ++p) {
      int c = p * 256 + tid;
      int row = c >> 3, col = (c & 7) * 8;
      *(uint4*)&a[row * LDA_S + col] = ar[p];
    }
#pragma unroll
    for (int p = 0; p < 4; ++p) {
      int c = p * 256 + tid;
      int row = c >> 4, col = (c & 15) * 8;
      *(uint4*)&b[row * LDB_S + col] = br[p];
    }
  };
#endif

  // consume one LDS buffer: 2 K-steps x (2 A-frags x 4 B-frags) WMMAs
  auto compute = [&](int buf) {
    const bf16_t* a = As + buf * A_TILE;
    const bf16_t* b = Bs + buf * B_TILE;
#pragma unroll
    for (int s = 0; s < 2; ++s) {
      // A 16x32 frag (ISA 7.12.2): lane -> row lane&15;
      // regs 0..3 = K kb..kb+7, regs 4..7 = K kb+16..kb+23
      v16bf af[2];
#pragma unroll
      for (int i = 0; i < 2; ++i) {
        const bf16_t* p = &a[(wm + i * 16 + lr) * LDA_S + s * 32 + kb];
        union { v16bf v; uint4 q[2]; } u;
        u.q[0] = *(const uint4*)p;
        u.q[1] = *(const uint4*)(p + 16);
        af[i] = u.v;
      }
      // B 32x16 frag: lane -> N col lane&15; elems 0..7 = K kb..kb+7, 8..15 = K kb+16..kb+23
      v16bf bfr[4];
#pragma unroll
      for (int j = 0; j < 4; ++j) {
        const bf16_t* p = &b[(s * 32 + kb) * LDB_S + wn + j * 16 + lr];
        v16bf t;
#pragma unroll
        for (int e = 0; e < 8; ++e) {
          t[e]     = p[e * LDB_S];
          t[e + 8] = p[(e + 16) * LDB_S];
        }
        bfr[j] = t;
      }
#pragma unroll
      for (int i = 0; i < 2; ++i)
#pragma unroll
        for (int j = 0; j < 4; ++j)
          acc[i][j] = __builtin_amdgcn_wmma_f32_16x16x32_bf16(
              false, af[i], false, bfr[j], (short)0, acc[i][j], false, false);
    }
  };

  const int nT = K / BK;
#if USE_ASYNC_LDS
  issueTile(0, 0);
  tileFence();
  for (int t = 0; t < nT; ++t) {
    const int cb = t & 1;
    if (t + 1 < nT) issueTile((t + 1) * BK, cb ^ 1);
    compute(cb);
    if (t + 1 < nT) tileFence();
  }
#else
  loadG(0);
  storeL(0);
  __syncthreads();
  for (int t = 0; t < nT; ++t) {
    const int cb = t & 1;
    if (t + 1 < nT) loadG((t + 1) * BK);
    compute(cb);
    if (t + 1 < nT) {
      storeL(cb ^ 1);
      __syncthreads();
    }
  }
#endif

  // ---- epilogue: C/D layout: lane -> n = lane&15, m = reg + 8*(lane>=16) ----
  const int nloc = lane & 15;
  const int mh   = (lane >> 4) * 8;
#pragma unroll
  for (int i = 0; i < 2; ++i)
#pragma unroll
    for (int j = 0; j < 4; ++j)
#pragma unroll
      for (int e = 0; e < 8; ++e) {
        int m = bm + wm + i * 16 + mh + e;
        int n = bn + wn + j * 16 + nloc;
        float v = acc[i][j][e];
        if constexpr (HAS_BIAS) v += bias[n];
        if constexpr (GELU) v = 0.5f * v * (1.0f + erff(v * 0.70710678118654752f));
        size_t idx = (size_t)m * N + n;
        if constexpr (HAS_RADD) v += Radd[idx];
        if constexpr (OUT_F) Cf[idx] = v;
        if constexpr (OUT_B) Cb[idx] = (bf16_t)v;
      }
}

// ---------------------------------------------------------------------------
// LayerNorm (wave per row): Yb (bf16, nullable) / Yf (f32, nullable)
// ---------------------------------------------------------------------------
template <int D>
__global__ __launch_bounds__(256)
void k_layernorm(const float* __restrict__ X, const float* __restrict__ g,
                 const float* __restrict__ bta, bf16_t* __restrict__ Yb,
                 float* __restrict__ Yf) {
  constexpr int NPER = D / 32;
  int row  = blockIdx.x * 8 + (threadIdx.x >> 5);
  int lane = threadIdx.x & 31;
  const float* x = X + (size_t)row * D;
  float v[NPER];
  float s = 0.f;
#pragma unroll
  for (int i = 0; i < NPER; ++i) { v[i] = x[lane + i * 32]; s += v[i]; }
#pragma unroll
  for (int m = 16; m; m >>= 1) s += __shfl_xor(s, m, 32);
  float mean = s * (1.0f / D);
  float ss = 0.f;
#pragma unroll
  for (int i = 0; i < NPER; ++i) { float d = v[i] - mean; ss += d * d; }
#pragma unroll
  for (int m = 16; m; m >>= 1) ss += __shfl_xor(ss, m, 32);
  float r = rsqrtf(ss * (1.0f / D) + 1e-5f);
#pragma unroll
  for (int i = 0; i < NPER; ++i) {
    int c = lane + i * 32;
    float y = (v[i] - mean) * r * g[c] + bta[c];
    if (Yb) Yb[(size_t)row * D + c] = (bf16_t)y;
    if (Yf) Yf[(size_t)row * D + c] = y;
  }
}

// Encoder final: LN then L2-normalize * sqrt(D)  (D=256), bf16 out
__global__ __launch_bounds__(256)
void k_ln_l2(const float* __restrict__ X, const float* __restrict__ g,
             const float* __restrict__ bta, bf16_t* __restrict__ Yb) {
  constexpr int D = 256;
  int row  = blockIdx.x * 8 + (threadIdx.x >> 5);
  int lane = threadIdx.x & 31;
  const float* x = X + (size_t)row * D;
  float v[8];
  float s = 0.f;
#pragma unroll
  for (int i = 0; i < 8; ++i) { v[i] = x[lane + i * 32]; s += v[i]; }
#pragma unroll
  for (int m = 16; m; m >>= 1) s += __shfl_xor(s, m, 32);
  float mean = s * (1.0f / D);
  float ss = 0.f;
#pragma unroll
  for (int i = 0; i < 8; ++i) { float d = v[i] - mean; ss += d * d; }
#pragma unroll
  for (int m = 16; m; m >>= 1) ss += __shfl_xor(ss, m, 32);
  float r = rsqrtf(ss * (1.0f / D) + 1e-5f);
  float y[8];
  float q = 0.f;
#pragma unroll
  for (int i = 0; i < 8; ++i) {
    int c = lane + i * 32;
    y[i] = (v[i] - mean) * r * g[c] + bta[c];
    q += y[i] * y[i];
  }
#pragma unroll
  for (int m = 16; m; m >>= 1) q += __shfl_xor(q, m, 32);
  float sc = 16.0f / fmaxf(sqrtf(q), 1e-12f);   // sqrt(256)=16
#pragma unroll
  for (int i = 0; i < 8; ++i)
    Yb[(size_t)row * D + lane + i * 32] = (bf16_t)(y[i] * sc);
}

// ---------------------------------------------------------------------------
// Fused attention per (batch, head): RMS-q/k with gains, scale, optional mask
// on keys, softmax over keys, A @ V.  nq<=32, nk<=64, head dim 64.
// ---------------------------------------------------------------------------
__global__ __launch_bounds__(256)
void k_attn(const bf16_t* __restrict__ Q, int ldq,
            const bf16_t* __restrict__ Kp, int ldk,
            const bf16_t* __restrict__ Vp, int ldv,
            bf16_t* __restrict__ O, int ldo,
            const float* __restrict__ qg, const float* __restrict__ kg,
            const int* __restrict__ mask,
            int nq, int nk, int nheads) {
  __shared__ float qs[32][65];
  __shared__ float ks[64][65];
  __shared__ float vs[64][65];
  __shared__ float sa[8][64];
  const int b = blockIdx.x / nheads;
  const int h = blockIdx.x % nheads;
  const int wave = threadIdx.x >> 5, lane = threadIdx.x & 31;
  const float scale = 0.125f;                 // DH^-0.5, DH=64
  const float NEGMAX = -3.402823466e38f;

  // keys (RMS-norm * kg) and values into LDS
  for (int j = wave; j < nk; j += 8) {
    const bf16_t* kr = Kp + (size_t)(b * nk + j) * ldk + h * 64;
    float a0 = (float)kr[lane], a1 = (float)kr[lane + 32];
    float ss = a0 * a0 + a1 * a1;
#pragma unroll
    for (int m = 16; m; m >>= 1) ss += __shfl_xor(ss, m, 32);
    float inv = 1.0f / fmaxf(sqrtf(ss * (1.0f / 64.0f)), 1e-8f);
    ks[j][lane]      = a0 * inv * kg[lane];
    ks[j][lane + 32] = a1 * inv * kg[lane + 32];
    const bf16_t* vr = Vp + (size_t)(b * nk + j) * ldv + h * 64;
    vs[j][lane]      = (float)vr[lane];
    vs[j][lane + 32] = (float)vr[lane + 32];
  }
  // queries (RMS-norm * qg * scale) into LDS
  for (int i = wave; i < nq; i += 8) {
    const bf16_t* qr = Q + (size_t)(b * nq + i) * ldq + h * 64;
    float a0 = (float)qr[lane], a1 = (float)qr[lane + 32];
    float ss = a0 * a0 + a1 * a1;
#pragma unroll
    for (int m = 16; m; m >>= 1) ss += __shfl_xor(ss, m, 32);
    float inv = scale / fmaxf(sqrtf(ss * (1.0f / 64.0f)), 1e-8f);
    qs[i][lane]      = a0 * inv * qg[lane];
    qs[i][lane + 32] = a1 * inv * qg[lane + 32];
  }
  __syncthreads();

  // scores + softmax + A@V (wave per query row; lane covers keys j, j+32)
  for (int i = wave; i < nq; i += 8) {
    float s0 = 0.f, s1 = NEGMAX;
    for (int d = 0; d < 64; ++d) s0 += qs[i][d] * ks[lane][d];
    if (nk > 32) {
      s1 = 0.f;
      for (int d = 0; d < 64; ++d) s1 += qs[i][d] * ks[lane + 32][d];
    }
    if (mask) {
      if (mask[b * nk + lane] == 0) s0 = NEGMAX;
      if (nk > 32 && mask[b * nk + lane + 32] == 0) s1 = NEGMAX;
    }
    float mx = fmaxf(s0, s1);
#pragma unroll
    for (int m = 16; m; m >>= 1) mx = fmaxf(mx, __shfl_xor(mx, m, 32));
    float e0 = __expf(s0 - mx);
    float e1 = (nk > 32) ? __expf(s1 - mx) : 0.f;
    float sum = e0 + e1;
#pragma unroll
    for (int m = 16; m; m >>= 1) sum += __shfl_xor(sum, m, 32);
    float rs = 1.0f / sum;
    sa[wave][lane]      = e0 * rs;
    sa[wave][lane + 32] = e1 * rs;     // wave-private row; no barrier needed
    float o0 = 0.f, o1 = 0.f;
    for (int j = 0; j < nk; ++j) {
      float a = sa[wave][j];
      o0 += a * vs[j][lane];
      o1 += a * vs[j][lane + 32];
    }
    bf16_t* orow = O + (size_t)(b * nq + i) * ldo + h * 64;
    orow[lane]      = (bf16_t)o0;
    orow[lane + 32] = (bf16_t)o1;
  }
}

// ---------------------------------------------------------------------------
// Host driver
// ---------------------------------------------------------------------------
extern "C" void kernel_launch(void* const* d_in, const int* in_sizes, int n_in,
                              void* d_out, int out_size, void* d_ws, size_t ws_size,
                              hipStream_t stream) {
  (void)in_sizes; (void)n_in; (void)out_size; (void)ws_size;

  // ---- input leaves (sorted recursive dict flatten, 43 leaves) ----
  const int*   att_mask = (const int*)  d_in[0];   // (B,64) bool as int
  const float* dp_fnb   = (const float*)d_in[1];
  const float* dp_fng   = (const float*)d_in[2];
  const float* dp_ipW   = (const float*)d_in[3];   // (256,1024)
  const float* dp_ipb   = (const float*)d_in[4];
  const float* dl_W1    = (const float*)d_in[5];   // (3,1024,4096)
  const float* dl_W2    = (const float*)d_in[6];   // (3,4096,1024)
  const float* dl_Wk    = (const float*)d_in[7];   // (3,1024,1024)
  const float* dl_Wo    = (const float*)d_in[8];
  const float* dl_Wq    = (const float*)d_in[9];
  const float* dl_Wv    = (const float*)d_in[10];
  const float* dl_b1    = (const float*)d_in[11];  // (3,4096)
  const float* dl_b2    = (const float*)d_in[12];  // (3,1024)
  const float* dl_bo    = (const float*)d_in[13];  // (3,1024)
  const float* dl_ffb   = (const float*)d_in[14];
  const float* dl_ffg   = (const float*)d_in[15];
  const float* dl_kg    = (const float*)d_in[16];  // (3,64)
  const float* dl_nb    = (const float*)d_in[17];
  const float* dl_ng    = (const float*)d_in[18];
  const float* dl_qg    = (const float*)d_in[19];  // (3,64)
  const float* dp_pos   = (const float*)d_in[20];  // (32,1024)
  const float* ep_fnb   = (const float*)d_in[21];
  const float* ep_fng   = (const float*)d_in[22];
  const float* ep_lat0  = (const float*)d_in[23];  // (32,256)
  const float* ep_lat1  = (const float*)d_in[24];  // (32,256)
  const float* el_W1    = (const float*)d_in[25];  // (3,256,1024)
  const float* el_W2    = (const float*)d_in[26];  // (3,1024,256)
  const float* el_Wkv   = (const float*)d_in[27];  // (3,1024,2048)
  const float* el_Wo    = (const float*)d_in[28];  // (3,1024,256)
  const float* el_Wq    = (const float*)d_in[29];  // (3,256,1024)
  const float* el_b1    = (const float*)d_in[30];  // (3,1024)
  const float* el_b2    = (const float*)d_in[31];  // (3,256)
  const float* el_bo    = (const float*)d_in[32];  // (3,256)
  const float* el_ffb   = (const float*)d_in[33];
  const float* el_ffg   = (const float*)d_in[34];
  const float* el_kg    = (const float*)d_in[35];  // (3,64)
  const float* el_nlb   = (const float*)d_in[36];
  const float* el_nlg   = (const float*)d_in[37];
  const float* el_nxb   = (const float*)d_in[38];
  const float* el_nxg   = (const float*)d_in[39];
  const float* el_qg    = (const float*)d_in[40];  // (3,64)
  const float* ep_pos   = (const float*)d_in[41];  // (64,1024)
  const float* enc_x    = (const float*)d_in[42];  // (256,64,1024)

  const int B = 256, H = 16;
  const size_t RX = 16384;   // B*SEQ rows
  const size_t RL = 8192;    // B*NLAT rows
  const float posScale = 0.03125f;  // 1024^-0.5

  // ---- workspace bump allocator ----
  size_t cur = 0;
  auto alloc = [&](size_t bytes) -> void* {
    void* p = (char*)d_ws + cur;
    cur += (bytes + 255) & ~(size_t)255;
    return p;
  };

  // bf16 weight copies
  bf16_t* wb_eWq  = (bf16_t*)alloc((size_t)3 * 256 * 1024 * 2);
  bf16_t* wb_eWkv = (bf16_t*)alloc((size_t)3 * 1024 * 2048 * 2);
  bf16_t* wb_eWo  = (bf16_t*)alloc((size_t)3 * 1024 * 256 * 2);
  bf16_t* wb_eW1  = (bf16_t*)alloc((size_t)3 * 256 * 1024 * 2);
  bf16_t* wb_eW2  = (bf16_t*)alloc((size_t)3 * 1024 * 256 * 2);
  bf16_t* wb_ipW  = (bf16_t*)alloc((size_t)256 * 1024 * 2);
  bf16_t* wb_dWq  = (bf16_t*)alloc((size_t)3 * 1024 * 1024 * 2);
  bf16_t* wb_dWk  = (bf16_t*)alloc((size_t)3 * 1024 * 1024 * 2);
  bf16_t* wb_dWv  = (bf16_t*)alloc((size_t)3 * 1024 * 1024 * 2);
  bf16_t* wb_dWo  = (bf16_t*)alloc((size_t)3 * 1024 * 1024 * 2);
  bf16_t* wb_dW1  = (bf16_t*)alloc((size_t)3 * 1024 * 4096 * 2);
  bf16_t* wb_dW2  = (bf16_t*)alloc((size_t)3 * 4096 * 1024 * 2);

  // activations
  float*  x_f   = (float*) alloc(RX * 1024 * 4);          // encoder x (+pos)
  bf16_t* xn_b  = (bf16_t*)alloc(RX * 1024 * 2);          // LN output (enc & dec)
  bf16_t* kv_b  = (bf16_t*)alloc(RX * 2048 * 2);          // enc K|V; dec q/k/v alias
  bf16_t* dq_b  = kv_b;
  bf16_t* dk_b  = kv_b + RL * 1024;
  bf16_t* dv_b  = kv_b + 2 * RL * 1024;
  bf16_t* q1_b  = (bf16_t*)alloc(RL * 1024 * 2);
  bf16_t* q0_b  = (bf16_t*)alloc(RL * 1024 * 2);
  bf16_t* o1_b  = (bf16_t*)alloc(RL * 1024 * 2);          // attn out (also dec)
  bf16_t* o0_b  = (bf16_t*)alloc(RL * 1024 * 2);
  float*  lc1   = (float*) alloc(RL * 256 * 4);
  float*  lc0   = (float*) alloc(RL * 256 * 4);
  bf16_t* lcn_b = (bf16_t*)alloc(RL * 256 * 2);
  bf16_t* ffh_b = (bf16_t*)alloc(RL * 4096 * 2);          // FF hidden (enc uses 1024)
  bf16_t* lat_b = (bf16_t*)alloc(RL * 256 * 2);
  float*  dx_f  = (float*) alloc(RL * 1024 * 4);          // decoder x

  // ---- launch helpers ----
  auto conv = [&](const float* s, bf16_t* d, size_t n) {
    k_f32_to_bf16<<<dim3(2048), dim3(256), 0, stream>>>(s, d, n);
  };
  // epilogue variants used:
  //   plain -> bf16           (projections)
  //   bias+gelu -> bf16       (FF hidden)
  //   bias+residual -> f32    (output projections / FF out)
  //   bias -> f32             (decoder input projection)
  auto gemm = [&](const bf16_t* A, const bf16_t* Bw, const float* bias,
                  const float* Radd, float* Cf, bf16_t* Cb,
                  int M, int N, int K, int gelu) {
    dim3 g(N / BN, M / BM), blk(256);
    if (Cb && !bias)
      k_gemm<false, false, false, false, true><<<g, blk, 0, stream>>>(A, Bw, bias, Radd, Cf, Cb, M, N, K);
    else if (Cb && bias && gelu)
      k_gemm<true, true, false, false, true><<<g, blk, 0, stream>>>(A, Bw, bias, Radd, Cf, Cb, M, N, K);
    else if (Cf && bias && Radd)
      k_gemm<true, false, true, true, false><<<g, blk, 0, stream>>>(A, Bw, bias, Radd, Cf, Cb, M, N, K);
    else
      k_gemm<true, false, false, true, false><<<g, blk, 0, stream>>>(A, Bw, bias, Radd, Cf, Cb, M, N, K);
  };
  auto ln1024 = [&](const float* X, const float* g, const float* b,
                    bf16_t* Yb, float* Yf, int rows) {
    k_layernorm<1024><<<dim3(rows / 8), dim3(256), 0, stream>>>(X, g, b, Yb, Yf);
  };
  auto ln256 = [&](const float* X, const float* g, const float* b,
                   bf16_t* Yb, float* Yf, int rows) {
    k_layernorm<256><<<dim3(rows / 8), dim3(256), 0, stream>>>(X, g, b, Yb, Yf);
  };
  auto attn = [&](const bf16_t* Q, int ldq, const bf16_t* Kp, int ldk,
                  const bf16_t* Vp, int ldv, bf16_t* O, int ldo,
                  const float* qg, const float* kg, const int* msk, int nq, int nk) {
    k_attn<<<dim3(B * H), dim3(256), 0, stream>>>(Q, ldq, Kp, ldk, Vp, ldv, O, ldo,
                                                  qg, kg, msk, nq, nk, H);
  };

  // ---- weight conversions (f32 -> bf16) ----
  conv(el_Wq,  wb_eWq,  (size_t)3 * 256 * 1024);
  conv(el_Wkv, wb_eWkv, (size_t)3 * 1024 * 2048);
  conv(el_Wo,  wb_eWo,  (size_t)3 * 1024 * 256);
  conv(el_W1,  wb_eW1,  (size_t)3 * 256 * 1024);
  conv(el_W2,  wb_eW2,  (size_t)3 * 1024 * 256);
  conv(dp_ipW, wb_ipW,  (size_t)256 * 1024);
  conv(dl_Wq,  wb_dWq,  (size_t)3 * 1024 * 1024);
  conv(dl_Wk,  wb_dWk,  (size_t)3 * 1024 * 1024);
  conv(dl_Wv,  wb_dWv,  (size_t)3 * 1024 * 1024);
  conv(dl_Wo,  wb_dWo,  (size_t)3 * 1024 * 1024);
  conv(dl_W1,  wb_dW1,  (size_t)3 * 1024 * 4096);
  conv(dl_W2,  wb_dW2,  (size_t)3 * 4096 * 1024);

  // ---- encoder setup: x = enc_x + pos * DLM^-0.5 ; broadcast latents ----
  k_add_pos<<<dim3(4096), dim3(256), 0, stream>>>(enc_x, ep_pos, x_f,
                                                  RX * 1024, (size_t)64 * 1024, posScale);
  k_bcast<<<dim3(1024), dim3(256), 0, stream>>>(ep_lat1, lc1, RL * 256, (size_t)32 * 256);
  k_bcast<<<dim3(1024), dim3(256), 0, stream>>>(ep_lat0, lc0, RL * 256, (size_t)32 * 256);

  // ---- encoder layers ----
  for (int l = 0; l < 3; ++l) {
    // xn = LN(x); KV = xn @ Wkv
    ln1024(x_f, el_nxg + l * 1024, el_nxb + l * 1024, xn_b, nullptr, (int)RX);
    gemm(xn_b, wb_eWkv + (size_t)l * 1024 * 2048, nullptr, nullptr, nullptr, kv_b,
         (int)RX, 2048, 1024, 0);
    // q1 = LN(lc1) @ Wq ; q0 = LN(lc0) @ Wq
    ln256(lc1, el_nlg + l * 256, el_nlb + l * 256, lcn_b, nullptr, (int)RL);
    gemm(lcn_b, wb_eWq + (size_t)l * 256 * 1024, nullptr, nullptr, nullptr, q1_b,
         (int)RL, 1024, 256, 0);
    ln256(lc0, el_nlg + l * 256, el_nlb + l * 256, lcn_b, nullptr, (int)RL);
    gemm(lcn_b, wb_eWq + (size_t)l * 256 * 1024, nullptr, nullptr, nullptr, q0_b,
         (int)RL, 1024, 256, 0);
    // attention (mask on c1 stream only)
    attn(q1_b, 1024, kv_b, 2048, kv_b + 1024, 2048, o1_b, 1024,
         el_qg + l * 64, el_kg + l * 64, att_mask, 32, 64);
    attn(q0_b, 1024, kv_b, 2048, kv_b + 1024, 2048, o0_b, 1024,
         el_qg + l * 64, el_kg + l * 64, nullptr, 32, 64);
    // lc += o @ Wo + bo
    gemm(o1_b, wb_eWo + (size_t)l * 1024 * 256, el_bo + l * 256, lc1, lc1, nullptr,
         (int)RL, 256, 1024, 0);
    gemm(o0_b, wb_eWo + (size_t)l * 1024 * 256, el_bo + l * 256, lc0, lc0, nullptr,
         (int)RL, 256, 1024, 0);
    // FF: lc += W2(gelu(W1(LN(lc))))
    ln256(lc1, el_ffg + l * 256, el_ffb + l * 256, lcn_b, nullptr, (int)RL);
    gemm(lcn_b, wb_eW1 + (size_t)l * 256 * 1024, el_b1 + l * 1024, nullptr, nullptr,
         ffh_b, (int)RL, 1024, 256, 1);
    gemm(ffh_b, wb_eW2 + (size_t)l * 1024 * 256, el_b2 + l * 256, lc1, lc1, nullptr,
         (int)RL, 256, 1024, 0);
    ln256(lc0, el_ffg + l * 256, el_ffb + l * 256, lcn_b, nullptr, (int)RL);
    gemm(lcn_b, wb_eW1 + (size_t)l * 256 * 1024, el_b1 + l * 1024, nullptr, nullptr,
         ffh_b, (int)RL, 1024, 256, 1);
    gemm(ffh_b, wb_eW2 + (size_t)l * 1024 * 256, el_b2 + l * 256, lc0, lc0, nullptr,
         (int)RL, 256, 1024, 0);
  }

  // ---- encoder final: LN + L2-normalize (c1 stream only: sentence_only) ----
  k_ln_l2<<<dim3(RL / 8), dim3(256), 0, stream>>>(lc1, ep_fng, ep_fnb, lat_b);

  // ---- decoder input: x = lat @ ipW + ipb ; += pos * DLM^-0.5 ----
  gemm(lat_b, wb_ipW, dp_ipb, nullptr, dx_f, nullptr, (int)RL, 1024, 256, 0);
  k_add_pos<<<dim3(4096), dim3(256), 0, stream>>>(dx_f, dp_pos, dx_f,
                                                  RL * 1024, (size_t)32 * 1024, posScale);

  // ---- decoder layers ----
  for (int l = 0; l < 3; ++l) {
    ln1024(dx_f, dl_ng + l * 1024, dl_nb + l * 1024, xn_b, nullptr, (int)RL);
    gemm(xn_b, wb_dWq + (size_t)l * 1024 * 1024, nullptr, nullptr, nullptr, dq_b,
         (int)RL, 1024, 1024, 0);
    gemm(xn_b, wb_dWk + (size_t)l * 1024 * 1024, nullptr, nullptr, nullptr, dk_b,
         (int)RL, 1024, 1024, 0);
    gemm(xn_b, wb_dWv + (size_t)l * 1024 * 1024, nullptr, nullptr, nullptr, dv_b,
         (int)RL, 1024, 1024, 0);
    attn(dq_b, 1024, dk_b, 1024, dv_b, 1024, o1_b, 1024,
         dl_qg + l * 64, dl_kg + l * 64, nullptr, 32, 32);
    gemm(o1_b, wb_dWo + (size_t)l * 1024 * 1024, dl_bo + l * 1024, dx_f, dx_f, nullptr,
         (int)RL, 1024, 1024, 0);
    ln1024(dx_f, dl_ffg + l * 1024, dl_ffb + l * 1024, xn_b, nullptr, (int)RL);
    gemm(xn_b, wb_dW1 + (size_t)l * 1024 * 4096, dl_b1 + l * 4096, nullptr, nullptr,
         ffh_b, (int)RL, 4096, 1024, 1);
    gemm(ffh_b, wb_dW2 + (size_t)l * 4096 * 1024, dl_b2 + l * 1024, dx_f, dx_f, nullptr,
         (int)RL, 1024, 4096, 0);
  }

  // ---- final LN -> d_out (f32, (B,32,1024)) ----
  ln1024(dx_f, dp_fng, dp_fnb, nullptr, (float*)d_out, (int)RL);
}
